// WindowAttention_773094113651
// MI455X (gfx1250) — matrix operand order repaired and checked by reading, not compiled, not run on hardware
//
#include <hip/hip_runtime.h>

#define DIMC 192
#define TDIM 576
#define NHEAD 6
#define HDIM 32
#define TOK 64
#define SHIFTV 4

typedef __attribute__((ext_vector_type(16))) _Float16 v16h;
typedef __attribute__((ext_vector_type(8)))  float    v8f;

union V16U { uint4 q[2]; v16h v; };

// A-fragment (16x32 f16): per lane 32 contiguous bytes along K.
__device__ __forceinline__ v16h frag_a(const _Float16* p) {
  V16U r;
  r.q[0] = *(const uint4*)(p);
  r.q[1] = *(const uint4*)(p + 8);
  return r.v;
}
// B-fragment (32x16 f16): per lane 32 contiguous bytes along K.
__device__ __forceinline__ v16h frag_b(const _Float16* p) {
  V16U r;
  r.q[0] = *(const uint4*)(p);
  r.q[1] = *(const uint4*)(p + 8);
  return r.v;
}
__device__ __forceinline__ v8f wmma16(v16h a, v16h b, v8f c) {
  return __builtin_amdgcn_wmma_f32_16x16x32_f16(false, a, false, b, (short)0, c,
                                                false, false);
}

// ---------------------------------------------------------------------------
// Prep: convert + transpose weights to f16, K-contiguous per output column.
//   qkvT[n][k] = qkv_w[k][n]  (576 x 192)
//   projT[n][k] = proj_w[k][n] (192 x 192)
// ---------------------------------------------------------------------------
__global__ void prep_weights(const float* __restrict__ qkv_w,
                             const float* __restrict__ proj_w,
                             _Float16* __restrict__ qkvT,
                             _Float16* __restrict__ projT) {
  int i = blockIdx.x * blockDim.x + threadIdx.x;
  if (i < TDIM * DIMC) {
    int n = i / DIMC, k = i % DIMC;
    qkvT[i] = (_Float16)qkv_w[k * TDIM + n];
  } else {
    int j = i - TDIM * DIMC;
    if (j < DIMC * DIMC) {
      int n = j / DIMC, k = j % DIMC;
      projT[j] = (_Float16)proj_w[k * DIMC + n];
    }
  }
}

// ---------------------------------------------------------------------------
// One workgroup (384 threads = 12 waves) per 8x8 window. Whole window pipeline
// (qkv GEMM -> scores -> softmax -> P@V -> proj) stays in LDS.
// ---------------------------------------------------------------------------
#define LDS_X     0          // Xf16   64x192            24576 B
#define LDS_Q     24576      // Q      64x192            24576 B
#define LDS_K     49152      // K      64x192            24576 B
#define LDS_VT    73728      // V^T   192x64             24576 B
#define LDS_OA    98304      // attn-out 64x192          24576 B
#define LDS_SF    122880     // scores f32 6x64x(65)     99840 B (stride 65: bank-spread)
#define LDS_PB    222720     // probs f16  6x64x64       49152 B
#define LDS_TOTAL 271872

__global__ __launch_bounds__(384, 1) void win_attn(
    const float* __restrict__ x, const _Float16* __restrict__ qkvT,
    const _Float16* __restrict__ projT, const float* __restrict__ qkv_b,
    const float* __restrict__ proj_b, float* __restrict__ out) {
  extern __shared__ char smem[];
  _Float16* Xh  = (_Float16*)(smem + LDS_X);
  _Float16* Qb  = (_Float16*)(smem + LDS_Q);
  _Float16* Kb  = (_Float16*)(smem + LDS_K);
  _Float16* VT  = (_Float16*)(smem + LDS_VT);
  _Float16* OA  = (_Float16*)(smem + LDS_OA);
  float*    SF  = (float*)   (smem + LDS_SF);
  _Float16* PB  = (_Float16*)(smem + LDS_PB);

  const int tid = threadIdx.x;
  const int W   = tid >> 5;        // wave id 0..11
  const int l   = tid & 31;
  const int ln  = l & 15;
  const int hiL = l >> 4;          // 0 for lanes 0-15, 1 for 16-31

  const int wid  = blockIdx.x;     // 4096 windows
  const int bb   = wid >> 10;
  const int widx = wid & 1023;
  const int wy   = widx >> 5;
  const int wx   = widx & 31;

  // ---- Phase 1: gather shifted window, f32 -> f16 into LDS --------------
  for (int i = tid; i < TOK * DIMC; i += 384) {
    int c = i >> 6, t = i & 63;
    int hh = (wy * 8 + (t >> 3) + SHIFTV) & 255;
    int ww = (wx * 8 + (t & 7) + SHIFTV) & 255;
    float v = x[(((size_t)bb * DIMC + c) << 16) + (hh << 8) + ww];
    Xh[t * DIMC + c] = (_Float16)v;
  }

  // ---- Phase 2: qkv = X(64x192) @ Wqkv(192x576) + b ---------------------
  // wave W owns N-tiles [3W, 3W+3), all 4 M-tiles; K in 6 steps of 32.
  // B fragments are software-pipelined: k-step 0 is issued BEFORE the
  // barrier (independent of LDS), k-step ks+1 is prefetched during ks.
  {
    const _Float16* bptr[3];
    v16h bcur[3];
    #pragma unroll
    for (int nt = 0; nt < 3; ++nt) {
      bptr[nt] = qkvT + ((W * 3 + nt) * 16 + ln) * DIMC + hiL * 16;
      bcur[nt] = frag_b(bptr[nt]);          // k-step 0 in flight across barrier
    }
    __syncthreads();

    v8f acc[4][3];
    #pragma unroll
    for (int nt = 0; nt < 3; ++nt) {
      float bv = qkv_b[(W * 3 + nt) * 16 + ln];
      #pragma unroll
      for (int mt = 0; mt < 4; ++mt)
        #pragma unroll
        for (int r = 0; r < 8; ++r) acc[mt][nt][r] = bv;
    }
    #pragma unroll
    for (int ks = 0; ks < 6; ++ks) {
      int kb = ks * 32;
      v16h bnxt[3];
      #pragma unroll
      for (int nt = 0; nt < 3; ++nt)
        bnxt[nt] = (ks < 5) ? frag_b(bptr[nt] + kb + 32) : bcur[nt];
      v16h af[4];
      #pragma unroll
      for (int mt = 0; mt < 4; ++mt)
        af[mt] = frag_a(Xh + (mt * 16 + ln) * DIMC + kb + hiL * 16);
      #pragma unroll
      for (int nt = 0; nt < 3; ++nt)
        #pragma unroll
        for (int mt = 0; mt < 4; ++mt)
          acc[mt][nt] = wmma16(af[mt], bcur[nt], acc[mt][nt]);
      #pragma unroll
      for (int nt = 0; nt < 3; ++nt) bcur[nt] = bnxt[nt];
    }
    #pragma unroll
    for (int nt = 0; nt < 3; ++nt) {
      int n = (W * 3 + nt) * 16 + ln;
      #pragma unroll
      for (int mt = 0; mt < 4; ++mt)
        #pragma unroll
        for (int r = 0; r < 8; ++r) {
          int m = mt * 16 + r + hiL * 8;
          _Float16 hv = (_Float16)acc[mt][nt][r];
          if (n < DIMC)            Qb[m * DIMC + n] = hv;
          else if (n < 2 * DIMC)   Kb[m * DIMC + (n - DIMC)] = hv;
          else                     VT[(n - 2 * DIMC) * TOK + m] = hv;   // transposed V
        }
    }
  }
  __syncthreads();

  // ---- Phase 3: scores = Q Kt / sqrt(32); 2 waves per head --------------
  {
    const int h = W >> 1, half = W & 1;
    const float scale = 0.17677669529663687f;
    #pragma unroll
    for (int mt2 = 0; mt2 < 2; ++mt2) {
      int mt = half * 2 + mt2;
      v16h af = frag_a(Qb + (mt * 16 + ln) * DIMC + h * HDIM + hiL * 16);
      #pragma unroll
      for (int nt = 0; nt < 4; ++nt) {
        v16h bf = frag_b(Kb + (nt * 16 + ln) * DIMC + h * HDIM + hiL * 16);
        v8f c = {};
        c = wmma16(af, bf, c);
        #pragma unroll
        for (int r = 0; r < 8; ++r)
          SF[h * (64 * 65) + (mt * 16 + r + hiL * 8) * 65 + nt * 16 + ln] =
              c[r] * scale;
      }
    }
  }
  __syncthreads();

  // ---- Phase 4: softmax, one of the 384 rows per thread -----------------
  // exp() cached back into the (thread-private) score row: 64 exps, not 128.
  {
    int h = tid / 64, m = tid & 63;
    float* rp = SF + h * (64 * 65) + m * 65;
    float mx = -3.4e38f;
    for (int j = 0; j < 64; ++j) mx = fmaxf(mx, rp[j]);
    float s = 0.f;
    for (int j = 0; j < 64; ++j) {
      float e = __expf(rp[j] - mx);
      s += e;
      rp[j] = e;
    }
    float inv = 1.0f / s;
    _Float16* op = PB + h * 4096 + m * 64;
    for (int j = 0; j < 64; ++j) op[j] = (_Float16)(rp[j] * inv);
  }
  __syncthreads();

  // ---- Phase 5: out_h = P(64x64) @ V(64x32); 2 waves per head -----------
  {
    const int h = W >> 1, half = W & 1;
    #pragma unroll
    for (int mt2 = 0; mt2 < 2; ++mt2) {
      int mt = half * 2 + mt2;
      const _Float16* ap = PB + h * 4096 + (mt * 16 + ln) * 64;
      v16h a0 = frag_a(ap + hiL * 16);
      v16h a1 = frag_a(ap + 32 + hiL * 16);
      #pragma unroll
      for (int nt = 0; nt < 2; ++nt) {
        const _Float16* vp = VT + (h * HDIM + nt * 16 + ln) * TOK;
        v16h b0 = frag_b(vp + hiL * 16);
        v16h b1 = frag_b(vp + 32 + hiL * 16);
        v8f c = {};
        c = wmma16(a0, b0, c);
        c = wmma16(a1, b1, c);
        #pragma unroll
        for (int r = 0; r < 8; ++r)
          OA[(mt * 16 + r + hiL * 8) * DIMC + h * HDIM + nt * 16 + ln] =
              (_Float16)c[r];
      }
    }
  }

  // ---- Phase 6: proj GEMM + scatter to NCHW with inverse roll -----------
  {
    const int n = W * 16 + ln;       // wave W owns N-tile W (12 tiles = 192)
    const _Float16* pb = projT + n * DIMC + hiL * 16;
    v16h bcur = frag_b(pb);          // first proj B frag in flight across barrier
    __syncthreads();

    v8f acc2[4];
    float bv = proj_b[n];
    #pragma unroll
    for (int mt = 0; mt < 4; ++mt)
      #pragma unroll
      for (int r = 0; r < 8; ++r) acc2[mt][r] = bv;
    #pragma unroll
    for (int ks = 0; ks < 6; ++ks) {
      int kb = ks * 32;
      v16h bnxt = (ks < 5) ? frag_b(pb + kb + 32) : bcur;
      #pragma unroll
      for (int mt = 0; mt < 4; ++mt) {
        v16h af = frag_a(OA + (mt * 16 + ln) * DIMC + kb + hiL * 16);
        acc2[mt] = wmma16(af, bcur, acc2[mt]);
      }
      bcur = bnxt;
    }
    #pragma unroll
    for (int mt = 0; mt < 4; ++mt)
      #pragma unroll
      for (int r = 0; r < 8; ++r) {
        int m = mt * 16 + r + hiL * 8;
        int hh = (wy * 8 + (m >> 3) + SHIFTV) & 255;
        int ww = (wx * 8 + (m & 7) + SHIFTV) & 255;
        out[(((size_t)bb * DIMC + n) << 16) + (hh << 8) + ww] = acc2[mt][r];
      }
  }
}

extern "C" void kernel_launch(void* const* d_in, const int* in_sizes, int n_in,
                              void* d_out, int out_size, void* d_ws, size_t ws_size,
                              hipStream_t stream) {
  const float* x      = (const float*)d_in[0];
  const float* qkv_w  = (const float*)d_in[1];
  const float* qkv_b  = (const float*)d_in[2];
  const float* proj_w = (const float*)d_in[3];
  const float* proj_b = (const float*)d_in[4];

  _Float16* qkvT  = (_Float16*)d_ws;                 // 576*192 f16
  _Float16* projT = qkvT + TDIM * DIMC;              // 192*192 f16

  // Convert/transpose weights once per launch (hot in 192MB L2 across 4096 WGs).
  int welems = TDIM * DIMC + DIMC * DIMC;            // 147456
  prep_weights<<<(welems + 255) / 256, 256, 0, stream>>>(qkv_w, proj_w, qkvT, projT);

  hipFuncSetAttribute((const void*)win_attn,
                      hipFuncAttributeMaxDynamicSharedMemorySize, LDS_TOTAL);

  // 4 batches * 32*32 windows = 4096 workgroups, 384 threads (12 waves) each.
  win_attn<<<4096, 384, LDS_TOTAL, stream>>>(x, qkvT, projT, qkv_b, proj_b,
                                             (float*)d_out);
}
